// decoder_884763263702
// MI455X (gfx1250) — matrix operand order repaired and checked by reading, not compiled, not run on hardware
//
// MI455X (gfx1250) attention-LSTM decoder.
// Strategy (bandwidth roofline): all per-step GEMMs have M=64 -> AI = 64 FLOP per
// weight element = 32 FLOP/B at bf16, deeply bandwidth bound. Convert every weight
// matrix + the attention input + keys_enc to bf16 once (~162 MB total), which fits
// the 192 MB L2, so the 256 sequential decode steps stream weights from L2 instead
// of HBM (23.3 TB/s). All matmuls use v_wmma_f32_16x16x32_bf16 with f32 accum.
// Requires ws_size >= ~180 MB.

#include <hip/hip_runtime.h>
#include <hip/hip_bf16.h>

#define B_  64
#define T_  256
#define C_  1024
#define D_  1024
#define V_  4096
#define E_  512

typedef __attribute__((ext_vector_type(16))) __bf16 bf16x16;
typedef __attribute__((ext_vector_type(8)))  float  f32x8;

union FragAB { bf16x16 v; uint4 q[2]; };

__device__ __forceinline__ float bf2f(__hip_bfloat16 x) { return __bfloat162float(x); }
__device__ __forceinline__ __hip_bfloat16 f2bf(float x) { return __float2bfloat16(x); }
__device__ __forceinline__ float sigf(float x) { return 1.0f / (1.0f + __expf(-x)); }

// ---------------------------------------------------------------------------
// WMMA GEMM:  out(MxN) = X(MxK) * W(NxK)^T [+ bias] [+ existing out]
// Row-major bf16 X and W (both K-contiguous, matching the x @ W.T reference).
// One wave computes a 16(M) x 32(N) tile; A fragment reused across 2 wmmas.
// Fragment loads use contiguous k order (k = 16*(lane/16) + s), same for A and B:
//   A 16x32: lane<16 holds M=lane, K=0..15; lane>=16 holds M=lane-16, K=16..31
//   B 32x16: lane<16 holds N=lane, K=0..15; lane>=16 holds N=lane-16, K=16..31
//   C/D    : acc[r] = C[r + (lane>=16 ? 8 : 0)][lane & 15]
// M%16==0, N%32==0, K%32==0, and total waves % 8 == 0 for all call sites.
// ---------------------------------------------------------------------------
__global__ __launch_bounds__(256) void k_gemm_bf16(
    const __hip_bfloat16* __restrict__ X, const __hip_bfloat16* __restrict__ W,
    const float* __restrict__ bias, float* __restrict__ outF, int outLd,
    __hip_bfloat16* __restrict__ outBf, int M, int N, int K, int accumulate)
{
  const int lane   = threadIdx.x & 31;
  const int wid    = blockIdx.x * (blockDim.x >> 5) + (threadIdx.x >> 5);
  const int tilesN = N >> 5;
  const int tm     = (wid / tilesN) << 4;
  const int tn     = (wid % tilesN) << 5;
  if (tm >= M) return;

  const int  lm = lane & 15;
  const bool hi = lane >= 16;

  const unsigned short* Xr = reinterpret_cast<const unsigned short*>(X)
                             + (size_t)(tm + lm) * K + (hi ? 16 : 0);
  const unsigned short* W0 = reinterpret_cast<const unsigned short*>(W)
                             + (size_t)(tn + lm) * K + (hi ? 16 : 0);
  const unsigned short* W1 = W0 + (size_t)16 * K;

  f32x8 acc0 = {};
  f32x8 acc1 = {};

  for (int k0 = 0; k0 < K; k0 += 32) {
    FragAB a, b0, b1;
    a.q[0]  = *reinterpret_cast<const uint4*>(Xr + k0);
    a.q[1]  = *reinterpret_cast<const uint4*>(Xr + k0 + 8);
    b0.q[0] = *reinterpret_cast<const uint4*>(W0 + k0);
    b0.q[1] = *reinterpret_cast<const uint4*>(W0 + k0 + 8);
    b1.q[0] = *reinterpret_cast<const uint4*>(W1 + k0);
    b1.q[1] = *reinterpret_cast<const uint4*>(W1 + k0 + 8);
    if (k0 + 160 <= K) {                       // pull next weight lines toward L0
      __builtin_prefetch(W0 + k0 + 128, 0, 1); // -> global_prefetch_b8
      __builtin_prefetch(W1 + k0 + 128, 0, 1);
    }
    acc0 = __builtin_amdgcn_wmma_f32_16x16x32_bf16(false, a.v, false, b0.v,
                                                   (short)0, acc0, false, false);
    acc1 = __builtin_amdgcn_wmma_f32_16x16x32_bf16(false, a.v, false, b1.v,
                                                   (short)0, acc1, false, false);
  }

  const int cn0 = tn + lm, cn1 = tn + 16 + lm;
  const int rb  = tm + (hi ? 8 : 0);
  const float bs0 = bias ? bias[cn0] : 0.0f;
  const float bs1 = bias ? bias[cn1] : 0.0f;
#pragma unroll
  for (int r = 0; r < 8; ++r) {
    const int row = rb + r;
    float v0 = acc0[r] + bs0;
    float v1 = acc1[r] + bs1;
    if (accumulate) {
      v0 += outF[(size_t)row * outLd + cn0];
      v1 += outF[(size_t)row * outLd + cn1];
    }
    if (outF) {
      outF[(size_t)row * outLd + cn0] = v0;
      outF[(size_t)row * outLd + cn1] = v1;
    }
    if (outBf) {
      outBf[(size_t)row * N + cn0] = f2bf(v0);
      outBf[(size_t)row * N + cn1] = f2bf(v1);
    }
  }
}

// ---------------------------------------------------------------------------
// Elementwise / reduction kernels
// ---------------------------------------------------------------------------
__global__ void k_cvt_bf16(const float* __restrict__ s, __hip_bfloat16* __restrict__ d,
                           size_t n) {
  size_t stride = (size_t)gridDim.x * blockDim.x;
  for (size_t i = (size_t)blockIdx.x * blockDim.x + threadIdx.x; i < n; i += stride)
    d[i] = f2bf(s[i]);
}

__global__ void k_add_f32(const float* __restrict__ a, const float* __restrict__ b,
                          float* __restrict__ d, int n) {
  int i = blockIdx.x * blockDim.x + threadIdx.x;
  if (i < n) d[i] = a[i] + b[i];
}

__global__ void k_fill_f32(float* p, size_t n, float v) {
  size_t stride = (size_t)gridDim.x * blockDim.x;
  for (size_t i = (size_t)blockIdx.x * blockDim.x + threadIdx.x; i < n; i += stride)
    p[i] = v;
}
__global__ void k_fill_bf16(__hip_bfloat16* p, size_t n, float v) {
  size_t stride = (size_t)gridDim.x * blockDim.x;
  for (size_t i = (size_t)blockIdx.x * blockDim.x + threadIdx.x; i < n; i += stride)
    p[i] = f2bf(v);
}
__global__ void k_fill_i32(int* p, size_t n, int v) {
  size_t stride = (size_t)gridDim.x * blockDim.x;
  for (size_t i = (size_t)blockIdx.x * blockDim.x + threadIdx.x; i < n; i += stride)
    p[i] = v;
}

// x_cat(B,2E) bf16 = [embd[prev[b]], attn[b]]
__global__ __launch_bounds__(256) void k_build_x(const int* __restrict__ prev,
                                                 const float* __restrict__ embd,
                                                 const float* __restrict__ attn,
                                                 __hip_bfloat16* __restrict__ x) {
  int idx = blockIdx.x * 256 + threadIdx.x;           // B*E threads
  int b = idx >> 9, j = idx & 511;
  x[(size_t)b * 1024 + j]       = f2bf(embd[(size_t)prev[b] * E_ + j]);
  x[(size_t)b * 1024 + 512 + j] = f2bf(attn[(size_t)b * E_ + j]);
}

// gates(B,4D) packed [i,f,g,o] -> update c (f32), write h (bf16)
__global__ __launch_bounds__(256) void k_lstm_update(const float* __restrict__ gates,
                                                     float* __restrict__ c,
                                                     __hip_bfloat16* __restrict__ h) {
  int idx = blockIdx.x * 256 + threadIdx.x;           // B*D threads
  int b = idx >> 10, j = idx & 1023;
  const float* g = gates + (size_t)b * 4 * D_;
  float i  = sigf(g[j]);
  float f  = sigf(g[D_ + j]);
  float gg = tanhf(g[2 * D_ + j]);
  float o  = sigf(g[3 * D_ + j]);
  float cn = f * c[idx] + i * gg;
  c[idx] = cn;
  h[idx] = f2bf(o * tanhf(cn));
}

// e[b,t] = sum_e v[e] * tanh(keys[b,t,e] + q[b,e]) ; one wave per (b,t)
__global__ __launch_bounds__(256) void k_scores(const __hip_bfloat16* __restrict__ keys,
                                                const float* __restrict__ q,
                                                const float* __restrict__ vv,
                                                float* __restrict__ e) {
  int lane = threadIdx.x & 31;
  int wid  = blockIdx.x * 8 + (threadIdx.x >> 5);
  int b = wid >> 8, t = wid & 255;
  const __hip_bfloat16* kr = keys + ((size_t)b * T_ + t) * E_;
  const float*          qr = q + (size_t)b * E_;
  float s = 0.0f;
  for (int i = lane; i < E_; i += 32) s += vv[i] * tanhf(bf2f(kr[i]) + qr[i]);
  for (int off = 16; off; off >>= 1) s += __shfl_xor(s, off, 32);
  if (lane == 0) e[(size_t)b * T_ + t] = s;
}

__global__ __launch_bounds__(256) void k_softmax_T(float* __restrict__ e) {
  __shared__ float sm[256];
  int b = blockIdx.x, tid = threadIdx.x;
  float x = e[(size_t)b * T_ + tid];
  sm[tid] = x; __syncthreads();
  for (int off = 128; off; off >>= 1) {
    if (tid < off) sm[tid] = fmaxf(sm[tid], sm[tid + off]);
    __syncthreads();
  }
  float m = sm[0]; __syncthreads();
  float ex = __expf(x - m);
  sm[tid] = ex; __syncthreads();
  for (int off = 128; off; off >>= 1) {
    if (tid < off) sm[tid] += sm[tid + off];
    __syncthreads();
  }
  e[(size_t)b * T_ + tid] = ex / sm[0];
}

// ctx[b,c] = sum_t a[b,t] * input[b,t,c]  (bf16 in/out)
__global__ __launch_bounds__(256) void k_ctx(const float* __restrict__ a,
                                             const __hip_bfloat16* __restrict__ xin,
                                             __hip_bfloat16* __restrict__ ctx) {
  __shared__ float sa[256];
  int b = blockIdx.x, tid = threadIdx.x;
  sa[tid] = a[(size_t)b * T_ + tid]; __syncthreads();
  float a0 = 0, a1 = 0, a2 = 0, a3 = 0;
  const __hip_bfloat16* base = xin + (size_t)b * T_ * C_;
  for (int t = 0; t < T_; ++t) {
    float w = sa[t];
    const __hip_bfloat16* row = base + (size_t)t * C_;
    a0 += w * bf2f(row[tid]);
    a1 += w * bf2f(row[tid + 256]);
    a2 += w * bf2f(row[tid + 512]);
    a3 += w * bf2f(row[tid + 768]);
  }
  ctx[(size_t)b * C_ + tid]       = f2bf(a0);
  ctx[(size_t)b * C_ + tid + 256] = f2bf(a1);
  ctx[(size_t)b * C_ + tid + 512] = f2bf(a2);
  ctx[(size_t)b * C_ + tid + 768] = f2bf(a3);
}

// layernorm + affine + relu over 2V=8192, bf16 output
__global__ __launch_bounds__(256) void k_ln_relu(const float* __restrict__ x,
                                                 const float* __restrict__ g,
                                                 const float* __restrict__ be,
                                                 __hip_bfloat16* __restrict__ y) {
  const int NN = 2 * V_;
  __shared__ float sm[256];
  int b = blockIdx.x, tid = threadIdx.x;
  float s = 0, ss = 0;
  for (int i = tid; i < NN; i += 256) {
    float v = x[(size_t)b * NN + i];
    s += v; ss += v * v;
  }
  sm[tid] = s; __syncthreads();
  for (int off = 128; off; off >>= 1) { if (tid < off) sm[tid] += sm[tid + off]; __syncthreads(); }
  s = sm[0]; __syncthreads();
  sm[tid] = ss; __syncthreads();
  for (int off = 128; off; off >>= 1) { if (tid < off) sm[tid] += sm[tid + off]; __syncthreads(); }
  ss = sm[0]; __syncthreads();
  float mu = s / NN;
  float rs = rsqrtf(ss / NN - mu * mu + 1e-5f);
  for (int i = tid; i < NN; i += 256) {
    float v = (x[(size_t)b * NN + i] - mu) * rs * g[i] + be[i];
    y[(size_t)b * NN + i] = f2bf(fmaxf(v, 0.0f));
  }
}

// Gumbel-max categorical sampling (deterministic hash noise)
__global__ __launch_bounds__(256) void k_sample(const float* __restrict__ logits,
                                                int ldRow, int t, int* __restrict__ prev) {
  __shared__ float sv[256];
  __shared__ int   si[256];
  int b = blockIdx.x, tid = threadIdx.x;
  float best = -3.4e38f; int bi = 0;
  for (int v = tid; v < V_; v += 256) {
    unsigned h = (unsigned)(t * 1315423911u) ^ (unsigned)(b * 2654435761u)
               ^ (unsigned)(v * 2246822519u);
    h ^= h >> 16; h *= 0x7feb352du; h ^= h >> 15; h *= 0x846ca68bu; h ^= h >> 16;
    float u   = (float)(h | 1u) * 2.3283064e-10f;
    float val = logits[(size_t)b * ldRow + v] + (-__logf(-__logf(u)));
    if (val > best) { best = val; bi = v; }
  }
  sv[tid] = best; si[tid] = bi; __syncthreads();
  for (int off = 128; off; off >>= 1) {
    if (tid < off && sv[tid + off] > sv[tid]) { sv[tid] = sv[tid + off]; si[tid] = si[tid + off]; }
    __syncthreads();
  }
  if (tid == 0) prev[b] = si[0];
}

// ---------------------------------------------------------------------------
extern "C" void kernel_launch(void* const* d_in, const int* in_sizes, int n_in,
                              void* d_out, int out_size, void* d_ws, size_t ws_size,
                              hipStream_t stream) {
  (void)in_sizes; (void)n_in; (void)out_size; (void)ws_size;
  typedef __hip_bfloat16 bf16;

  const float* in_x = (const float*)d_in[0];
  const float* embd = (const float*)d_in[1];
  // lstm leaf order per cell(): for g in {I,F,C,O}: i_w, i_b, h_w, h_b
  const float *l1_iw[4], *l1_ib[4], *l1_hw[4], *l1_hb[4];
  const float *l2_iw[4], *l2_ib[4], *l2_hw[4], *l2_hb[4];
  for (int g = 0; g < 4; ++g) {
    l1_iw[g] = (const float*)d_in[2 + 4 * g + 0];
    l1_ib[g] = (const float*)d_in[2 + 4 * g + 1];
    l1_hw[g] = (const float*)d_in[2 + 4 * g + 2];
    l1_hb[g] = (const float*)d_in[2 + 4 * g + 3];
    l2_iw[g] = (const float*)d_in[18 + 4 * g + 0];
    l2_ib[g] = (const float*)d_in[18 + 4 * g + 1];
    l2_hw[g] = (const float*)d_in[18 + 4 * g + 2];
    l2_hb[g] = (const float*)d_in[18 + 4 * g + 3];
  }
  const float* enc_w = (const float*)d_in[34];
  const float* enc_b = (const float*)d_in[35];
  const float* dec_w = (const float*)d_in[36];
  const float* dec_b = (const float*)d_in[37];
  const float* att_v = (const float*)d_in[38];
  const float* out_w = (const float*)d_in[39];
  const float* out_b = (const float*)d_in[40];
  const float* w1    = (const float*)d_in[41];
  const float* b1    = (const float*)d_in[42];
  const float* ln_g  = (const float*)d_in[43];
  const float* ln_be = (const float*)d_in[44];
  const float* w2    = (const float*)d_in[45];
  const float* b2    = (const float*)d_in[46];
  float* out = (float*)d_out;

  // ---- workspace layout (<= ~180 MB) ----
  char* ws = (char*)d_ws;
  size_t off = 0;
  auto alloc = [&](size_t bytes) -> void* {
    void* p = ws + off; off = (off + bytes + 255) & ~(size_t)255; return p;
  };
  bf16* W1i  = (bf16*)alloc((size_t)4 * D_ * 1024 * 2);   // lstm1 input gates, packed
  bf16* W1h  = (bf16*)alloc((size_t)4 * D_ * D_ * 2);
  bf16* W2i  = (bf16*)alloc((size_t)4 * D_ * D_ * 2);
  bf16* W2h  = (bf16*)alloc((size_t)4 * D_ * D_ * 2);
  bf16* ENC  = (bf16*)alloc((size_t)E_ * C_ * 2);
  bf16* DEC  = (bf16*)alloc((size_t)E_ * D_ * 2);
  bf16* OUTW = (bf16*)alloc((size_t)E_ * C_ * 2);
  bf16* MW1  = (bf16*)alloc((size_t)2 * V_ * D_ * 2);
  bf16* MW2  = (bf16*)alloc((size_t)V_ * 2 * V_ * 2);
  bf16* XINB = (bf16*)alloc((size_t)B_ * T_ * C_ * 2);
  bf16* KEYS = (bf16*)alloc((size_t)B_ * T_ * E_ * 2);
  bf16* H1   = (bf16*)alloc((size_t)B_ * D_ * 2);
  bf16* H2   = (bf16*)alloc((size_t)B_ * D_ * 2);
  bf16* X1   = (bf16*)alloc((size_t)B_ * 1024 * 2);
  bf16* CTX  = (bf16*)alloc((size_t)B_ * C_ * 2);
  bf16* HIDN = (bf16*)alloc((size_t)B_ * 2 * V_ * 2);
  float* B1S  = (float*)alloc((size_t)4 * D_ * 4);
  float* B2S  = (float*)alloc((size_t)4 * D_ * 4);
  float* Cc1  = (float*)alloc((size_t)B_ * D_ * 4);
  float* Cc2  = (float*)alloc((size_t)B_ * D_ * 4);
  float* G1   = (float*)alloc((size_t)B_ * 4 * D_ * 4);
  float* G2   = (float*)alloc((size_t)B_ * 4 * D_ * 4);
  float* Qb   = (float*)alloc((size_t)B_ * E_ * 4);
  float* SC   = (float*)alloc((size_t)B_ * T_ * 4);
  float* ATT  = (float*)alloc((size_t)B_ * E_ * 4);
  float* HID  = (float*)alloc((size_t)B_ * 2 * V_ * 4);
  int*   PREV = (int*)alloc((size_t)B_ * 4);

  auto cvt = [&](const float* s, bf16* d, size_t n) {
    int blocks = (int)((n + 255) / 256); if (blocks > 4096) blocks = 4096;
    k_cvt_bf16<<<blocks, 256, 0, stream>>>(s, d, n);
  };
  auto gemm = [&](const bf16* X, const bf16* W, const float* bias, float* oF, int ld,
                  bf16* oB, int M, int N, int K, int acc) {
    int blocks = (M / 16) * (N / 32) / 8;
    k_gemm_bf16<<<blocks, 256, 0, stream>>>(X, W, bias, oF, ld, oB, M, N, K, acc);
  };

  // ---- one-time setup: bf16 weight packs, fused biases, keys_enc ----
  for (int g = 0; g < 4; ++g) {
    const size_t m = (size_t)D_ * D_;  // all gate mats are 1024x1024 here
    cvt(l1_iw[g], W1i + g * m, m);
    cvt(l1_hw[g], W1h + g * m, m);
    cvt(l2_iw[g], W2i + g * m, m);
    cvt(l2_hw[g], W2h + g * m, m);
    k_add_f32<<<4, 256, 0, stream>>>(l1_ib[g], l1_hb[g], B1S + g * D_, D_);
    k_add_f32<<<4, 256, 0, stream>>>(l2_ib[g], l2_hb[g], B2S + g * D_, D_);
  }
  cvt(enc_w, ENC, (size_t)E_ * C_);
  cvt(dec_w, DEC, (size_t)E_ * D_);
  cvt(out_w, OUTW, (size_t)E_ * C_);
  cvt(w1, MW1, (size_t)2 * V_ * D_);
  cvt(w2, MW2, (size_t)V_ * 2 * V_);
  cvt(in_x, XINB, (size_t)B_ * T_ * C_);
  // keys_enc (B*T, E) = input @ enc_w^T + enc_b, stored bf16
  gemm(XINB, ENC, enc_b, nullptr, 0, KEYS, B_ * T_, E_, C_, 0);

  // ---- zero initial state ----
  k_fill_bf16<<<64, 256, 0, stream>>>(H1, (size_t)B_ * D_, 0.0f);
  k_fill_bf16<<<64, 256, 0, stream>>>(H2, (size_t)B_ * D_, 0.0f);
  k_fill_f32<<<64, 256, 0, stream>>>(Cc1, (size_t)B_ * D_, 0.0f);
  k_fill_f32<<<64, 256, 0, stream>>>(Cc2, (size_t)B_ * D_, 0.0f);
  k_fill_f32<<<64, 256, 0, stream>>>(ATT, (size_t)B_ * E_, 0.0f);
  k_fill_i32<<<1, 64, 0, stream>>>(PREV, (size_t)B_, 0);

  // ---- sequential decode ----
  for (int t = 0; t < T_; ++t) {
    k_build_x<<<(B_ * E_) / 256, 256, 0, stream>>>(PREV, embd, ATT, X1);
    // LSTM1: gates = x @ Wi^T + (bi+bh) + h1 @ Wh^T
    gemm(X1, W1i, B1S, G1, 4 * D_, nullptr, B_, 4 * D_, 1024, 0);
    gemm(H1, W1h, nullptr, G1, 4 * D_, nullptr, B_, 4 * D_, D_, 1);
    k_lstm_update<<<(B_ * D_) / 256, 256, 0, stream>>>(G1, Cc1, H1);
    // LSTM2
    gemm(H1, W2i, B2S, G2, 4 * D_, nullptr, B_, 4 * D_, D_, 0);
    gemm(H2, W2h, nullptr, G2, 4 * D_, nullptr, B_, 4 * D_, D_, 1);
    k_lstm_update<<<(B_ * D_) / 256, 256, 0, stream>>>(G2, Cc2, H2);
    // attention
    gemm(H2, DEC, dec_b, Qb, E_, nullptr, B_, E_, D_, 0);
    k_scores<<<(B_ * T_) / 8, 256, 0, stream>>>(KEYS, Qb, att_v, SC);
    k_softmax_T<<<B_, 256, 0, stream>>>(SC);
    k_ctx<<<B_, 256, 0, stream>>>(SC, XINB, CTX);
    gemm(CTX, OUTW, out_b, ATT, E_, nullptr, B_, E_, C_, 0);
    // head -> logits straight into d_out (B,T,V)
    gemm(H2, MW1, b1, HID, 2 * V_, nullptr, B_, 2 * V_, D_, 0);
    k_ln_relu<<<B_, 256, 0, stream>>>(HID, ln_g, ln_be, HIDN);
    gemm(HIDN, MW2, b2, out + (size_t)t * V_, T_ * V_, nullptr, B_, V_, 2 * V_, 0);
    k_sample<<<B_, 256, 0, stream>>>(out + (size_t)t * V_, T_ * V_, t, PREV);
  }
}
